// EagerBidirectionalAttentionBlock_74509092651379
// MI455X (gfx1250) — matrix operand order repaired and checked by reading, not compiled, not run on hardware
//
#include <hip/hip_runtime.h>

// ---------------------------------------------------------------------------
// EagerBidirectionalAttentionBlock on MI455X (gfx1250)
// bf16 WMMA everywhere, f32 accumulate. Flash-style online-softmax attention.
// ---------------------------------------------------------------------------

#define HIDDEN 1024
#define NHEADS 16
#define HEADD  64
#define BATCH  2
#define SEQ    2048
#define MTOT   (BATCH * SEQ)   // 4096 rows for the projection GEMMs

typedef __attribute__((ext_vector_type(16))) __bf16 bf16x16;
typedef __attribute__((ext_vector_type(8)))  float  f32x8;

struct alignas(16) U128 { unsigned int w[4]; };
struct alignas(16) F128 { float f[4]; };

union Frag {
    bf16x16 bf;
    U128    q[2];
};

__device__ __forceinline__ unsigned short f32_to_bf16(float f) {
    union { float f; unsigned int u; } v; v.f = f;
    unsigned int r = v.u + 0x7FFFu + ((v.u >> 16) & 1u);  // round-to-nearest-even
    return (unsigned short)(r >> 16);
}

// ---------------------------------------------------------------------------
// fp32 -> bf16 bulk cast (n multiple of 1024)
// ---------------------------------------------------------------------------
__global__ __launch_bounds__(256) void cast_bf16_kernel(
    const float* __restrict__ src, unsigned short* __restrict__ dst, int n)
{
    int i = (blockIdx.x * 256 + threadIdx.x) * 4;
    if (i + 3 < n) {
        F128 f = *(const F128*)(src + i);
        dst[i + 0] = f32_to_bf16(f.f[0]);
        dst[i + 1] = f32_to_bf16(f.f[1]);
        dst[i + 2] = f32_to_bf16(f.f[2]);
        dst[i + 3] = f32_to_bf16(f.f[3]);
    }
}

// ---------------------------------------------------------------------------
// Y = X @ W^T + bias via v_wmma_f32_16x16x32_bf16.
//  X: [4096][1024] bf16 row-major, W: [1024][1024] bf16 row-major (torch Linear
//  weight: row n = output channel n over K). B-fragment column n == row n of W,
//  so all fragment loads are contiguous 16B.
//  mode 0: store bf16 Q[b,h,s,d]   mode 1: store bf16 K[b,h,s,d]
//  mode 2: store bf16 Vt[b,h,d,s]  mode 3: store f32  out[m][n]
//  Block: 256 thr (8 waves) -> 128x64 tile, wave -> 32x32 (2x2 WMMA tiles).
// ---------------------------------------------------------------------------
__global__ __launch_bounds__(256) void proj_gemm_kernel(
    const unsigned short* __restrict__ X,
    const unsigned short* __restrict__ W,
    const float* __restrict__ bias,
    unsigned short* __restrict__ out_bf,
    float* __restrict__ out_f,
    int mode)
{
    const int K = HIDDEN;
    const int lane = threadIdx.x & 31;
    const int wave = threadIdx.x >> 5;
    const int half = lane >> 4;
    const int ln   = lane & 15;

    const int m0 = blockIdx.y * 128 + (wave & 3) * 32;
    const int n0 = blockIdx.x * 64  + (wave >> 2) * 32;

    f32x8 acc[2][2] = {};

    for (int k0 = 0; k0 < K; k0 += 32) {
        Frag a[2], b[2];
#pragma unroll
        for (int i = 0; i < 2; ++i) {
            const unsigned short* ap = X + (size_t)(m0 + i * 16 + ln) * K + k0;
            a[i].q[0] = *(const U128*)(ap + half * 8);        // K = half*8 .. +7
            a[i].q[1] = *(const U128*)(ap + 16 + half * 8);   // K = 16+half*8 ..
        }
#pragma unroll
        for (int j = 0; j < 2; ++j) {
            const unsigned short* bp = W + (size_t)(n0 + j * 16 + ln) * K + k0;
            b[j].q[0] = *(const U128*)(bp + half * 8);        // K = half*8 .. +7
            b[j].q[1] = *(const U128*)(bp + 16 + half * 8);   // K = 16+half*8 ..
        }
#pragma unroll
        for (int i = 0; i < 2; ++i)
#pragma unroll
            for (int j = 0; j < 2; ++j)
                acc[i][j] = __builtin_amdgcn_wmma_f32_16x16x32_bf16(
                    false, a[i].bf, false, b[j].bf, (short)0, acc[i][j], false, false);
    }

    // Epilogue: D layout = VGPR v, row m = v + half*8, col n = ln.
#pragma unroll
    for (int i = 0; i < 2; ++i)
#pragma unroll
        for (int j = 0; j < 2; ++j) {
            const int n  = n0 + j * 16 + ln;
            const float bv = bias[n];
#pragma unroll
            for (int v = 0; v < 8; ++v) {
                const int   m   = m0 + i * 16 + half * 8 + v;
                const float val = acc[i][j][v] + bv;
                if (mode == 3) {
                    out_f[(size_t)m * HIDDEN + n] = val;
                } else {
                    const int bb = m >> 11, s = m & (SEQ - 1);
                    const int h  = n >> 6,  d = n & (HEADD - 1);
                    size_t addr;
                    if (mode == 2)  // Vt[b,h,d,s]
                        addr = ((size_t)(bb * NHEADS + h) * HEADD + d) * SEQ + s;
                    else            // Q/K [b,h,s,d]
                        addr = ((size_t)(bb * NHEADS + h) * SEQ + s) * HEADD + d;
                    out_bf[addr] = f32_to_bf16(val);
                }
            }
        }
}

// ---------------------------------------------------------------------------
// Flash-style attention. 128 thr (4 waves); wave owns 16 queries of (b,h).
// Streams keys 32 at a time: 4 score WMMAs + online softmax + 4 PV WMMAs.
// P goes D-layout -> LDS -> A-layout for the PV matmul.
// ---------------------------------------------------------------------------
__global__ __launch_bounds__(128) void attn_kernel(
    const unsigned short* __restrict__ Q,    // [B,H,S,64] bf16
    const unsigned short* __restrict__ Kb,   // [B,H,S,64] bf16
    const unsigned short* __restrict__ Vt,   // [B,H,64,S] bf16
    const unsigned char*  __restrict__ mask, // [B,S] bool
    unsigned short* __restrict__ ctx)        // [B,S,1024] bf16
{
    __shared__ __align__(16) unsigned short plds[4][16][32];

    const int lane = threadIdx.x & 31;
    const int wave = threadIdx.x >> 5;
    const int half = lane >> 4;
    const int ln   = lane & 15;

    const int b  = blockIdx.z;
    const int h  = blockIdx.y;
    const int q0 = blockIdx.x * 64 + wave * 16;

    const unsigned short* Qh = Q  + ((size_t)(b * NHEADS + h) * SEQ + q0) * HEADD;
    const unsigned short* Kh = Kb + (size_t)(b * NHEADS + h) * SEQ * HEADD;
    const unsigned short* Vh = Vt + (size_t)(b * NHEADS + h) * HEADD * SEQ;

    // Preload Q A-fragments for the two K-dim (=d) chunks.
    Frag qa[2];
#pragma unroll
    for (int dd = 0; dd < 2; ++dd) {
        const unsigned short* qp = Qh + (size_t)ln * HEADD + dd * 32;
        qa[dd].q[0] = *(const U128*)(qp + half * 8);
        qa[dd].q[1] = *(const U128*)(qp + 16 + half * 8);
    }

    f32x8 cacc[4] = {};
    float rmax[8], rsum[8];
#pragma unroll
    for (int v = 0; v < 8; ++v) { rmax[v] = -3.0e38f; rsum[v] = 0.0f; }

    const float scale = 0.125f;  // 1/sqrt(64)

    for (int j0 = 0; j0 < SEQ; j0 += 32) {
        // ---- scores: two 16x16 D tiles over keys [j0..j0+31] ----
        f32x8 s0 = {}, s1 = {};
#pragma unroll
        for (int dd = 0; dd < 2; ++dd) {
            Frag kb;
            const unsigned short* kp = Kh + (size_t)(j0 + ln) * HEADD + dd * 32;
            kb.q[0] = *(const U128*)(kp + half * 8);
            kb.q[1] = *(const U128*)(kp + 16 + half * 8);
            s0 = __builtin_amdgcn_wmma_f32_16x16x32_bf16(
                false, qa[dd].bf, false, kb.bf, (short)0, s0, false, false);
        }
#pragma unroll
        for (int dd = 0; dd < 2; ++dd) {
            Frag kb;
            const unsigned short* kp = Kh + (size_t)(j0 + 16 + ln) * HEADD + dd * 32;
            kb.q[0] = *(const U128*)(kp + half * 8);
            kb.q[1] = *(const U128*)(kp + 16 + half * 8);
            s1 = __builtin_amdgcn_wmma_f32_16x16x32_bf16(
                false, qa[dd].bf, false, kb.bf, (short)0, s1, false, false);
        }

        // ---- scale + mask (column of s0 = ln, of s1 = ln+16) ----
        const float mk0 = mask[b * SEQ + j0 + ln]      ? 0.0f : -1.0e30f;
        const float mk1 = mask[b * SEQ + j0 + 16 + ln] ? 0.0f : -1.0e30f;

        float cmax[8];
#pragma unroll
        for (int v = 0; v < 8; ++v) {
            s0[v] = s0[v] * scale + mk0;
            s1[v] = s1[v] * scale + mk1;
            cmax[v] = fmaxf(s0[v], s1[v]);
        }
        // row max across the 16 lanes of each half (row = half*8 + v)
#pragma unroll
        for (int off = 1; off < 16; off <<= 1)
#pragma unroll
            for (int v = 0; v < 8; ++v)
                cmax[v] = fmaxf(cmax[v], __shfl_xor(cmax[v], off, 16));

        float alpha[8], csum[8];
#pragma unroll
        for (int v = 0; v < 8; ++v) {
            const float mnew = fmaxf(rmax[v], cmax[v]);
            alpha[v] = __expf(rmax[v] - mnew);
            rmax[v]  = mnew;
            const float p0 = __expf(s0[v] - mnew);
            const float p1 = __expf(s1[v] - mnew);
            csum[v] = p0 + p1;
            const int r = half * 8 + v;
            plds[wave][r][ln]      = f32_to_bf16(p0);
            plds[wave][r][ln + 16] = f32_to_bf16(p1);
        }
#pragma unroll
        for (int off = 1; off < 16; off <<= 1)
#pragma unroll
            for (int v = 0; v < 8; ++v)
                csum[v] += __shfl_xor(csum[v], off, 16);
#pragma unroll
        for (int v = 0; v < 8; ++v)
            rsum[v] = rsum[v] * alpha[v] + csum[v];

        // rescale running context accumulators
#pragma unroll
        for (int t = 0; t < 4; ++t)
#pragma unroll
            for (int v = 0; v < 8; ++v)
                cacc[t][v] *= alpha[v];

        // per-wave LDS region: same-wave DS ops are in-order, just drain DScnt
        asm volatile("s_wait_dscnt 0" ::: "memory");

        // ---- re-read P in A-fragment layout: row = ln, K chunks over j ----
        Frag pa;
        const unsigned short* pp = &plds[wave][ln][0];
        pa.q[0] = *(const U128*)(pp + half * 8);
        pa.q[1] = *(const U128*)(pp + 16 + half * 8);

        // ---- ctx += P @ V  (B column d = row d of Vt, contiguous in j) ----
#pragma unroll
        for (int t = 0; t < 4; ++t) {
            Frag vb;
            const unsigned short* vp = Vh + (size_t)(t * 16 + ln) * SEQ + j0;
            vb.q[0] = *(const U128*)(vp + half * 8);
            vb.q[1] = *(const U128*)(vp + 16 + half * 8);
            cacc[t] = __builtin_amdgcn_wmma_f32_16x16x32_bf16(
                false, pa.bf, false, vb.bf, (short)0, cacc[t], false, false);
        }
    }

    // ---- normalize + store ctx[b, s, h*64+d] as bf16 for the O-projection ----
#pragma unroll
    for (int t = 0; t < 4; ++t) {
        const int d = t * 16 + ln;
#pragma unroll
        for (int v = 0; v < 8; ++v) {
            const int   s   = q0 + half * 8 + v;
            const float val = cacc[t][v] / rsum[v];
            ctx[((size_t)(b * SEQ + s)) * HIDDEN + h * HEADD + d] = f32_to_bf16(val);
        }
    }
}

// ---------------------------------------------------------------------------
// Host-side orchestration
// ---------------------------------------------------------------------------
extern "C" void kernel_launch(void* const* d_in, const int* in_sizes, int n_in,
                              void* d_out, int out_size, void* d_ws, size_t ws_size,
                              hipStream_t stream)
{
    const float*         x    = (const float*)d_in[0];
    const unsigned char* mask = (const unsigned char*)d_in[1];
    const float* Wq_w = (const float*)d_in[2];
    const float* Wq_b = (const float*)d_in[3];
    const float* Wk_w = (const float*)d_in[4];
    const float* Wk_b = (const float*)d_in[5];
    const float* Wv_w = (const float*)d_in[6];
    const float* Wv_b = (const float*)d_in[7];
    const float* Wo_w = (const float*)d_in[8];
    const float* Wo_b = (const float*)d_in[9];
    float* out = (float*)d_out;

    char* ws = (char*)d_ws;
    const size_t MB = 1ull << 20;
    unsigned short* xb   = (unsigned short*)(ws + 0 * MB);   // 8 MB  x bf16
    unsigned short* wqb  = (unsigned short*)(ws + 8 * MB);   // 2 MB
    unsigned short* wkb  = (unsigned short*)(ws + 10 * MB);  // 2 MB
    unsigned short* wvb  = (unsigned short*)(ws + 12 * MB);  // 2 MB
    unsigned short* wob  = (unsigned short*)(ws + 14 * MB);  // 2 MB
    unsigned short* Qb   = (unsigned short*)(ws + 16 * MB);  // 8 MB [B,H,S,64]
    unsigned short* Kb   = (unsigned short*)(ws + 24 * MB);  // 8 MB [B,H,S,64]
    unsigned short* Vtb  = (unsigned short*)(ws + 32 * MB);  // 8 MB [B,H,64,S]
    unsigned short* ctxb = (unsigned short*)(ws + 40 * MB);  // 8 MB [B,S,1024]

    const int nx = MTOT * HIDDEN;     // 4M elements
    const int nw = HIDDEN * HIDDEN;   // 1M elements
    cast_bf16_kernel<<<nx / 1024, 256, 0, stream>>>(x,    xb,  nx);
    cast_bf16_kernel<<<nw / 1024, 256, 0, stream>>>(Wq_w, wqb, nw);
    cast_bf16_kernel<<<nw / 1024, 256, 0, stream>>>(Wk_w, wkb, nw);
    cast_bf16_kernel<<<nw / 1024, 256, 0, stream>>>(Wv_w, wvb, nw);
    cast_bf16_kernel<<<nw / 1024, 256, 0, stream>>>(Wo_w, wob, nw);

    dim3 pg(HIDDEN / 64, MTOT / 128);  // (16, 32)
    proj_gemm_kernel<<<pg, 256, 0, stream>>>(xb, wqb, Wq_b, Qb,  nullptr, 0);
    proj_gemm_kernel<<<pg, 256, 0, stream>>>(xb, wkb, Wk_b, Kb,  nullptr, 1);
    proj_gemm_kernel<<<pg, 256, 0, stream>>>(xb, wvb, Wv_b, Vtb, nullptr, 2);

    dim3 ag(SEQ / 64, NHEADS, BATCH);  // (32, 16, 2)
    attn_kernel<<<ag, 128, 0, stream>>>(Qb, Kb, Vtb, mask, ctxb);

    proj_gemm_kernel<<<pg, 256, 0, stream>>>(ctxb, wob, Wo_b, nullptr, out, 3);
}
